// RelativeAttention1D_50981261803707
// MI455X (gfx1250) — matrix operand order repaired and checked by reading, not compiled, not run on hardware
//
#include <hip/hip_runtime.h>
#include <hip/hip_bf16.h>

// ---------------------------------------------------------------------------
// RelativeAttention1D on MI455X (gfx1250, wave32, WMMA)
// B=16, N=1024, INP=OUP=512, H=8, Dh=64.
// bf16 WMMA (f32 accumulate) everywhere; flash-style attention so the
// 512 MB N x N score tensor never touches HBM.
// Data movement: GLOBAL_LOAD_ASYNC_TO_LDS_B128 (ASYNCcnt) for bf16 tiles,
// double-buffered in the output GEMM; global_prefetch for the f32 X stream.
// Workspace: 4 x 512KiB weightsT (bf16) + 4 x 16MiB Q/K/V/Ctx (bf16) ~ 66MiB.
// ---------------------------------------------------------------------------

typedef __attribute__((ext_vector_type(16))) __bf16 v16bf;
typedef __attribute__((ext_vector_type(8)))  float  v8f;

union FragBF { uint4 q[2]; v16bf v; };

__device__ __forceinline__ unsigned short f2bf(float f) {
    unsigned u = __float_as_uint(f);
    unsigned r = (u + 0x7FFFu + ((u >> 16) & 1u)) >> 16;   // RNE
    return (unsigned short)r;
}

// Async memory->LDS copy, 16 bytes per lane. GVS addressing:
// mem = saddr(SGPR64) + voff(VGPR32) ; LDS dest = lds base-relative byte addr.
// Tracked on ASYNCcnt; fence with async_wait() before the consuming barrier.
__device__ __forceinline__ void async_b128(unsigned lds_byte_addr,
                                           unsigned voff_bytes,
                                           unsigned long long sbase) {
    asm volatile("global_load_async_to_lds_b128 %0, %1, %2"
                 :: "v"(lds_byte_addr), "v"(voff_bytes), "s"(sbase)
                 : "memory");
}
__device__ __forceinline__ void async_wait() {
    asm volatile("s_wait_asynccnt 0x0" ::: "memory");
}
__device__ __forceinline__ unsigned lds_addr(const void* p) {
    return (unsigned)(size_t)p;   // generic LDS ptr: low 32 bits = LDS offset
}

constexpr int BQ     = 16;
constexpr int NSEQ   = 1024;
constexpr int DINP   = 512;
constexpr int DINNER = 512;
constexpr int NH     = 8;
constexpr int DH     = 64;
constexpr int ROWS   = BQ * NSEQ;      // 16384

constexpr int TM = 128, TN = 64, TK = 32;
constexpr int XS_STRIDE = 40;          // 32 + 8 pad (bf16 elems) -> bank spread
constexpr int WT_STRIDE = 40;

// ---------------------------------------------------------------------------
// Kernel 1: W[k][n] f32 -> W^T[n][k] bf16
// ---------------------------------------------------------------------------
__global__ __launch_bounds__(256) void k_wt(const float* __restrict__ W,
                                            unsigned short* __restrict__ Wt) {
    int idx = blockIdx.x * 256 + threadIdx.x;      // 0 .. 512*512-1
    int n = idx >> 9, k = idx & 511;
    Wt[idx] = f2bf(W[k * 512 + n]);
}

// ---------------------------------------------------------------------------
// Kernel 2: Y[b][h][n][d] (bf16) = X[row][k] (f32) @ W (via W^T bf16)
// block: 256 thr (8 waves); tile 128(M) x 64(N); K-steps of 32.
// W tile arrives via async-to-LDS; X tile is converted f32->bf16 in VALU.
// ---------------------------------------------------------------------------
__global__ __launch_bounds__(256) void k_proj_gemm(
    const float* __restrict__ X, const unsigned short* __restrict__ Wt,
    unsigned short* __restrict__ Y)
{
    __shared__ __align__(16) unsigned short xs[TM * XS_STRIDE];
    __shared__ __align__(16) unsigned short wt[TN * WT_STRIDE];

    const int tid  = threadIdx.x;
    const int wave = tid >> 5, lane = tid & 31;
    const int rowBase = blockIdx.x * TM;
    const int nBase   = blockIdx.y * TN;
    const int lane15  = lane & 15;

    v8f acc[4] = {};

    for (int kt = 0; kt < DINP; kt += TK) {
        __syncthreads();
        // W^T tile: 64 rows x 32 k, async memory->LDS (one b128 per thread)
        {
            int nloc = tid >> 2, c8 = tid & 3;
            async_b128(lds_addr(&wt[nloc * WT_STRIDE + c8 * 8]),
                       (unsigned)((nloc * DINP + c8 * 8) * 2),
                       (unsigned long long)(Wt + (size_t)nBase * DINP + kt));
        }
        // X tile: 128x32 f32 -> bf16 LDS (manual: needs conversion)
        for (int e = tid; e < TM * TK / 4; e += 256) {
            int r = e >> 3, c4 = e & 7;
            const float* src = X + (size_t)(rowBase + r) * DINP + kt + c4 * 4;
            const float4 xv = *(const float4*)src;
            if (kt + TK < DINP) __builtin_prefetch(src + TK, 0, 0);
            uint2 pk;
            pk.x = (unsigned)f2bf(xv.x) | ((unsigned)f2bf(xv.y) << 16);
            pk.y = (unsigned)f2bf(xv.z) | ((unsigned)f2bf(xv.w) << 16);
            *(uint2*)(&xs[r * XS_STRIDE + c4 * 4]) = pk;
        }
        async_wait();
        __syncthreads();

        const int mloc = wave * 16 + lane15;
        const int aklo = (lane < 16) ? 0 : 8;
        FragBF a;
        a.q[0] = *(const uint4*)(&xs[mloc * XS_STRIDE + aklo]);
        a.q[1] = *(const uint4*)(&xs[mloc * XS_STRIDE + 16 + aklo]);
        const int bklo = (lane < 16) ? 0 : 8;
#pragma unroll
        for (int c = 0; c < 4; ++c) {
            int ncl = c * 16 + lane15;
            FragBF bfr;
            bfr.q[0] = *(const uint4*)(&wt[ncl * WT_STRIDE + bklo]);
            bfr.q[1] = *(const uint4*)(&wt[ncl * WT_STRIDE + bklo + 16]);
            acc[c] = __builtin_amdgcn_wmma_f32_16x16x32_bf16(
                false, a.v, false, bfr.v, (short)0, acc[c], false, false);
        }
    }

    const int mhalf = (lane < 16) ? 0 : 8;
#pragma unroll
    for (int c = 0; c < 4; ++c) {
        int colg = nBase + c * 16 + lane15;
        int h = colg >> 6, d = colg & 63;
#pragma unroll
        for (int j = 0; j < 8; ++j) {
            int rowg = rowBase + wave * 16 + j + mhalf;
            int b = rowg >> 10, n = rowg & 1023;
            Y[(size_t)(((b * NH + h) * NSEQ) + n) * DH + d] = f2bf(acc[c][j]);
        }
    }
}

// ---------------------------------------------------------------------------
// Kernel 3: flash attention. grid (N/128, H, B); block 256 (8 waves).
// Wave owns 16 query rows; streams keys in tiles of 32 (K tile async->LDS).
// ---------------------------------------------------------------------------
__global__ __launch_bounds__(256) void k_attn(
    const unsigned short* __restrict__ Q, const unsigned short* __restrict__ K,
    const unsigned short* __restrict__ V, const float* __restrict__ bias_table,
    unsigned short* __restrict__ Ctx)
{
    __shared__ __align__(16) unsigned short ks[32 * 72];   // K tile [key][d]
    __shared__ __align__(16) unsigned short vt[64 * 40];   // V tile transposed [d][key]
    __shared__ __align__(16) float          lbias[1152];   // bias window over i-j
    __shared__ __align__(16) unsigned short pw[8][16 * 40];// per-wave P scratch

    const int tid  = threadIdx.x;
    const int wave = tid >> 5, lane = tid & 31;
    const int lane15 = lane & 15;
    const int qt = blockIdx.x, h = blockIdx.y, b = blockIdx.z;
    const int qb0 = qt * 128;
    const size_t bh = (size_t)(b * NH + h);
    const unsigned short* Qp = Q + bh * NSEQ * DH;
    const unsigned short* Kp = K + bh * NSEQ * DH;
    const unsigned short* Vp = V + bh * NSEQ * DH;

    // bias window: indices (i - j + 1023) for i in [qb0, qb0+127], j in [0,1023]
    for (int t = tid; t < 1151; t += 256)
        lbias[t] = bias_table[(qb0 + t) * NH + h];

    // Q fragments (d 0..31 and 32..63), kept in registers for whole key loop
    const int qrow = qb0 + wave * 16 + lane15;
    const int aklo = (lane < 16) ? 0 : 8;
    FragBF qa0, qa1;
    qa0.q[0] = *(const uint4*)(Qp + (size_t)qrow * DH + aklo);
    qa0.q[1] = *(const uint4*)(Qp + (size_t)qrow * DH + 16 + aklo);
    qa1.q[0] = *(const uint4*)(Qp + (size_t)qrow * DH + 32 + aklo);
    qa1.q[1] = *(const uint4*)(Qp + (size_t)qrow * DH + 48 + aklo);

    float mstat[8], lstat[8];
#pragma unroll
    for (int j = 0; j < 8; ++j) { mstat[j] = -1e30f; lstat[j] = 0.0f; }
    v8f o[4] = {};

    const float scale = 0.125f;   // 64^-0.5
    const int mhalf = (lane < 16) ? 0 : 8;
    const int bklo  = (lane < 16) ? 0 : 8;

    for (int kt = 0; kt < NSEQ; kt += 32) {
        __syncthreads();
        const int keyloc = tid >> 3, c8 = tid & 7;
        // K tile: async memory->LDS, row-major (one b128 per thread)
        async_b128(lds_addr(&ks[keyloc * 72 + c8 * 8]),
                   (unsigned)((keyloc * DH + c8 * 8) * 2),
                   (unsigned long long)(Kp + (size_t)kt * DH));
        // V tile: manual transpose into LDS
        {
            uint4 vv = *(const uint4*)(Vp + (size_t)(kt + keyloc) * DH + c8 * 8);
            const unsigned short* vs = (const unsigned short*)&vv;
#pragma unroll
            for (int t = 0; t < 8; ++t)
                vt[(c8 * 8 + t) * 40 + keyloc] = vs[t];
        }
        async_wait();
        __syncthreads();

        // scores: two 16-key halves, K-dim = d (2 chained WMMAs each)
        v8f s[2];
#pragma unroll
        for (int half = 0; half < 2; ++half) {
            int keyl = half * 16 + lane15;
            FragBF kb0, kb1;
            kb0.q[0] = *(const uint4*)(&ks[keyl * 72 + bklo]);
            kb0.q[1] = *(const uint4*)(&ks[keyl * 72 + bklo + 16]);
            kb1.q[0] = *(const uint4*)(&ks[keyl * 72 + 32 + bklo]);
            kb1.q[1] = *(const uint4*)(&ks[keyl * 72 + 32 + bklo + 16]);
            v8f z = {};
            z = __builtin_amdgcn_wmma_f32_16x16x32_bf16(
                false, qa0.v, false, kb0.v, (short)0, z, false, false);
            z = __builtin_amdgcn_wmma_f32_16x16x32_bf16(
                false, qa1.v, false, kb1.v, (short)0, z, false, false);
            s[half] = z;
        }

        // scale + relative bias + row-max (16-lane xor reductions)
        float rm[8];
#pragma unroll
        for (int j = 0; j < 8; ++j) {
            int iloc = wave * 16 + j + mhalf;
            int j0 = kt + lane15, j1 = kt + 16 + lane15;
            float s0 = s[0][j] * scale + lbias[iloc + 1023 - j0];
            float s1 = s[1][j] * scale + lbias[iloc + 1023 - j1];
            s[0][j] = s0; s[1][j] = s1;
            float r = fmaxf(s0, s1);
            r = fmaxf(r, __shfl_xor(r, 1));
            r = fmaxf(r, __shfl_xor(r, 2));
            r = fmaxf(r, __shfl_xor(r, 4));
            r = fmaxf(r, __shfl_xor(r, 8));
            rm[j] = r;
        }

        // online softmax update; stage P as bf16 in per-wave LDS scratch
        unsigned short* pwv = pw[wave];
        float fsc[8];
#pragma unroll
        for (int j = 0; j < 8; ++j) {
            float mnew = fmaxf(mstat[j], rm[j]);
            float f  = __expf(mstat[j] - mnew);
            float p0 = __expf(s[0][j] - mnew);
            float p1 = __expf(s[1][j] - mnew);
            float ps = p0 + p1;
            ps += __shfl_xor(ps, 1); ps += __shfl_xor(ps, 2);
            ps += __shfl_xor(ps, 4); ps += __shfl_xor(ps, 8);
            lstat[j] = lstat[j] * f + ps;
            mstat[j] = mnew;
            fsc[j] = f;
            int m = j + mhalf;
            pwv[m * 40 + lane15]      = f2bf(p0);
            pwv[m * 40 + 16 + lane15] = f2bf(p1);
        }
#pragma unroll
        for (int c = 0; c < 4; ++c)
#pragma unroll
            for (int j = 0; j < 8; ++j)
                o[c][j] *= fsc[j];

        // P(16x32) @ V(32x64): re-fragment P as A operand (wave-private LDS,
        // DS ops from the same wave are in-order -> no barrier needed)
        FragBF pa;
        pa.q[0] = *(const uint4*)(&pwv[lane15 * 40 + aklo]);
        pa.q[1] = *(const uint4*)(&pwv[lane15 * 40 + 16 + aklo]);
#pragma unroll
        for (int c = 0; c < 4; ++c) {
            int dcol = c * 16 + lane15;
            FragBF vb;
            vb.q[0] = *(const uint4*)(&vt[dcol * 40 + bklo]);
            vb.q[1] = *(const uint4*)(&vt[dcol * 40 + bklo + 16]);
            o[c] = __builtin_amdgcn_wmma_f32_16x16x32_bf16(
                false, pa.v, false, vb.v, (short)0, o[c], false, false);
        }
    }

    // finalize: divide by l, store ctx bf16 as [b][n][h*64+d]
#pragma unroll
    for (int j = 0; j < 8; ++j) {
        float inv = 1.0f / lstat[j];
        int n = qb0 + wave * 16 + j + mhalf;
#pragma unroll
        for (int c = 0; c < 4; ++c) {
            int col = h * DH + c * 16 + lane15;
            Ctx[((size_t)b * NSEQ + n) * DINNER + col] = f2bf(o[c][j] * inv);
        }
    }
}

// ---------------------------------------------------------------------------
// Kernel 4: out[row][col] f32 = Ctx(bf16) @ Wo + bo
// Fully async double-buffered: tile k+1 streams into LDS on ASYNCcnt while
// WMMAs consume tile k.
// ---------------------------------------------------------------------------
__global__ __launch_bounds__(256) void k_out_gemm(
    const unsigned short* __restrict__ Xb, const unsigned short* __restrict__ Wt,
    const float* __restrict__ bo, float* __restrict__ out)
{
    __shared__ __align__(16) unsigned short xs[2][TM * XS_STRIDE];
    __shared__ __align__(16) unsigned short wt[2][TN * WT_STRIDE];

    const int tid  = threadIdx.x;
    const int wave = tid >> 5, lane = tid & 31;
    const int rowBase = blockIdx.x * TM;
    const int nBase   = blockIdx.y * TN;
    const int lane15  = lane & 15;

    v8f acc[4] = {};

    auto issue = [&](int kt, int buf) {
        // X tile: 128x32 bf16 = 512 b128 chunks, 2 per thread
        for (int e = tid; e < TM * TK / 8; e += 256) {
            int r = e >> 2, c8 = e & 3;
            async_b128(lds_addr(&xs[buf][r * XS_STRIDE + c8 * 8]),
                       (unsigned)((r * DINNER + c8 * 8) * 2),
                       (unsigned long long)(Xb + (size_t)rowBase * DINNER + kt));
        }
        // W^T tile: 64x32 bf16, 1 b128 per thread
        int nloc = tid >> 2, c8 = tid & 3;
        async_b128(lds_addr(&wt[buf][nloc * WT_STRIDE + c8 * 8]),
                   (unsigned)((nloc * DINNER + c8 * 8) * 2),
                   (unsigned long long)(Wt + (size_t)nBase * DINNER + kt));
    };

    issue(0, 0);
    for (int kt = 0; kt < DINNER; kt += TK) {
        const int cur = (kt / TK) & 1;
        async_wait();          // tile `cur` fully resident in LDS (this wave)
        __syncthreads();       // ... and for all waves; `cur^1` free to fill
        if (kt + TK < DINNER) issue(kt + TK, cur ^ 1);

        const int mloc = wave * 16 + lane15;
        const int aklo = (lane < 16) ? 0 : 8;
        FragBF a;
        a.q[0] = *(const uint4*)(&xs[cur][mloc * XS_STRIDE + aklo]);
        a.q[1] = *(const uint4*)(&xs[cur][mloc * XS_STRIDE + 16 + aklo]);
        const int bklo = (lane < 16) ? 0 : 8;
#pragma unroll
        for (int c = 0; c < 4; ++c) {
            int ncl = c * 16 + lane15;
            FragBF bfr;
            bfr.q[0] = *(const uint4*)(&wt[cur][ncl * WT_STRIDE + bklo]);
            bfr.q[1] = *(const uint4*)(&wt[cur][ncl * WT_STRIDE + bklo + 16]);
            acc[c] = __builtin_amdgcn_wmma_f32_16x16x32_bf16(
                false, a.v, false, bfr.v, (short)0, acc[c], false, false);
        }
        __syncthreads();       // done reading `cur` before next fill targets it
    }

    const int mhalf = (lane < 16) ? 0 : 8;
#pragma unroll
    for (int c = 0; c < 4; ++c) {
        int colg = nBase + c * 16 + lane15;
        float bv = bo[colg];
#pragma unroll
        for (int j = 0; j < 8; ++j) {
            int rowg = rowBase + wave * 16 + j + mhalf;
            out[(size_t)rowg * DINNER + colg] = acc[c][j] + bv;
        }
    }
}

// ---------------------------------------------------------------------------
extern "C" void kernel_launch(void* const* d_in, const int* in_sizes, int n_in,
                              void* d_out, int out_size, void* d_ws, size_t ws_size,
                              hipStream_t stream) {
    const float* q_in  = (const float*)d_in[0];
    const float* k_in  = (const float*)d_in[1];
    const float* v_in  = (const float*)d_in[2];
    const float* Wq    = (const float*)d_in[3];
    const float* Wk    = (const float*)d_in[4];
    const float* Wv    = (const float*)d_in[5];
    const float* Wo    = (const float*)d_in[6];
    const float* bo    = (const float*)d_in[7];
    const float* btab  = (const float*)d_in[8];

    unsigned short* p   = (unsigned short*)d_ws;
    const size_t WSZ = 512 * 512;                 // one weight matrix (elems)
    const size_t TSZ = (size_t)ROWS * DINNER;     // one activation tensor (elems)
    unsigned short* Wqt = p;
    unsigned short* Wkt = Wqt + WSZ;
    unsigned short* Wvt = Wkt + WSZ;
    unsigned short* Wot = Wvt + WSZ;
    unsigned short* Qb  = Wot + WSZ;
    unsigned short* Kb  = Qb + TSZ;
    unsigned short* Vb  = Kb + TSZ;
    unsigned short* Ctx = Vb + TSZ;
    // required ws: (4*WSZ + 4*TSZ) * 2 bytes ~= 66 MiB

    k_wt<<<dim3(512 * 512 / 256), 256, 0, stream>>>(Wq, Wqt);
    k_wt<<<dim3(512 * 512 / 256), 256, 0, stream>>>(Wk, Wkt);
    k_wt<<<dim3(512 * 512 / 256), 256, 0, stream>>>(Wv, Wvt);
    k_wt<<<dim3(512 * 512 / 256), 256, 0, stream>>>(Wo, Wot);

    dim3 gg(ROWS / TM, DINNER / TN);
    k_proj_gemm<<<gg, 256, 0, stream>>>(q_in, Wqt, Qb);
    k_proj_gemm<<<gg, 256, 0, stream>>>(k_in, Wkt, Kb);
    k_proj_gemm<<<gg, 256, 0, stream>>>(v_in, Wvt, Vb);

    dim3 ga(NSEQ / 128, NH, BQ);
    k_attn<<<ga, 256, 0, stream>>>(Qb, Kb, Vb, btab, Ctx);

    k_out_gemm<<<gg, 256, 0, stream>>>(Ctx, Wot, bo, (float*)d_out);
}
